// DSSMamba_83708912599102
// MI455X (gfx1250) — matrix operand order repaired and checked by reading, not compiled, not run on hardware
//
#include <hip/hip_runtime.h>
#include <hip/hip_bf16.h>
#include <math.h>

typedef __attribute__((ext_vector_type(16))) _Float16 v16h;
typedef __attribute__((ext_vector_type(8)))  float    v8f;
typedef __attribute__((ext_vector_type(4)))  float    v4f;
typedef __attribute__((ext_vector_type(4)))  _Float16 v4h;
typedef __attribute__((ext_vector_type(2)))  __fp16   h2;   // return type of cvt_pkrtz

#define BB      16
#define LL      900
#define DMODEL  256
#define DINNER  512
#define NXZ     2048          // 4*DINNER
#define NROWS   (BB*LL)       // 14400
#define XDW     48            // dt_rank + 2*d_state
#define NST     16
#define BSTR    48            // padded column stride (halves) for LDS B tiles

__device__ __forceinline__ float siluf(float x)     { return x / (1.f + __expf(-x)); }
__device__ __forceinline__ float sigmf(float x)     { return 1.f / (1.f + __expf(-x)); }
__device__ __forceinline__ float softplusf(float x) { return (x > 20.f) ? x : log1pf(__expf(x)); }

// Pack 16 already-transformed floats (two contiguous 8-float runs at ap[0..7],
// ap[16..23]) into a WMMA A-fragment using packed f32->f16 converts.
__device__ __forceinline__ v16h pack_afrag(v4f x0, v4f x1, v4f x2, v4f x3) {
    union { v16h v; h2 p[8]; } u;
    u.p[0] = __builtin_amdgcn_cvt_pkrtz(x0[0], x0[1]);
    u.p[1] = __builtin_amdgcn_cvt_pkrtz(x0[2], x0[3]);
    u.p[2] = __builtin_amdgcn_cvt_pkrtz(x1[0], x1[1]);
    u.p[3] = __builtin_amdgcn_cvt_pkrtz(x1[2], x1[3]);
    u.p[4] = __builtin_amdgcn_cvt_pkrtz(x2[0], x2[1]);
    u.p[5] = __builtin_amdgcn_cvt_pkrtz(x2[2], x2[3]);
    u.p[6] = __builtin_amdgcn_cvt_pkrtz(x3[0], x3[1]);
    u.p[7] = __builtin_amdgcn_cvt_pkrtz(x3[2], x3[3]);
    return u.v;
}

// ---------------------------------------------------------------------------
// GEMM 1: XZ[14400,2048] = H[14400,256] @ Win[256,2048]
// block = 128 thr (4 waves); tile 64(M) x 64(N); B tile staged column-major
// f16 in LDS (stride 48 halves -> 32B-aligned columns, one v16h per fragment)
// ---------------------------------------------------------------------------
__global__ __launch_bounds__(128) void k_inproj(const float* __restrict__ H,
                                                const float* __restrict__ W,
                                                float* __restrict__ XZ) {
    __shared__ __attribute__((aligned(32))) _Float16 bt[64 * BSTR];
    const int tid  = threadIdx.x;
    const int wave = tid >> 5, lane = tid & 31;
    const int r    = lane & 15;
    const bool hi  = lane >= 16;
    const int m0   = blockIdx.y * 64 + wave * 16;
    const int n0   = blockIdx.x * 64;
    const int rowA = m0 + r;
    const int kb   = hi ? 8 : 0;
    const int roff = hi ? 16 : 0;
    const int sc   = tid & 63;            // staging column
    const int sg   = (tid >> 6) << 4;     // staging row base (0 / 16)
    const float* Wn = W + n0;

    v8f acc[4] = {};
    #pragma unroll
    for (int k0 = 0; k0 < 256; k0 += 32) {
        __syncthreads();
        #pragma unroll
        for (int it = 0; it < 4; ++it) {
            const int r0 = sg + it * 4;
            float f0 = Wn[(size_t)(k0 + r0 + 0) * 2048 + sc];
            float f1 = Wn[(size_t)(k0 + r0 + 1) * 2048 + sc];
            float f2 = Wn[(size_t)(k0 + r0 + 2) * 2048 + sc];
            float f3 = Wn[(size_t)(k0 + r0 + 3) * 2048 + sc];
            union { v4h v; h2 p[2]; } u;
            u.p[0] = __builtin_amdgcn_cvt_pkrtz(f0, f1);
            u.p[1] = __builtin_amdgcn_cvt_pkrtz(f2, f3);
            *(v4h*)(&bt[sc * BSTR + (((r0 & 8) << 1) | ((r0 & 16) >> 1) | (r0 & 7))]) = u.v;
        }
        __syncthreads();
        const float* ap = H + (size_t)rowA * 256 + k0 + kb;
        v16h a = pack_afrag(*(const v4f*)(ap),      *(const v4f*)(ap + 4),
                            *(const v4f*)(ap + 16), *(const v4f*)(ap + 20));
        #pragma unroll
        for (int t = 0; t < 4; ++t) {
            v16h bf = *(const v16h*)(&bt[(t * 16 + r) * BSTR + roff]);
            acc[t] = __builtin_amdgcn_wmma_f32_16x16x32_f16(false, a, false, bf,
                                                            (short)0, acc[t], false, false);
        }
    }
    #pragma unroll
    for (int t = 0; t < 4; ++t)
        #pragma unroll
        for (int j = 0; j < 8; ++j)
            XZ[(size_t)(m0 + j + (hi ? 8 : 0)) * 2048 + n0 + t * 16 + r] = acc[t][j];
}

// ---------------------------------------------------------------------------
// GEMM 2: XD[br][14400,48] = silu(x_br[14400,512]) @ Wx_br[512,48]
// ---------------------------------------------------------------------------
__global__ __launch_bounds__(128) void k_xd(const float* __restrict__ XZ,
                                            const float* __restrict__ Wf,
                                            const float* __restrict__ Wb,
                                            float* __restrict__ XDf,
                                            float* __restrict__ XDb) {
    __shared__ __attribute__((aligned(32))) _Float16 bt[48 * BSTR];
    const int br = blockIdx.y;
    const float* W  = br ? Wb  : Wf;
    float*       XD = br ? XDb : XDf;
    const float* A  = XZ + br * 1024;     // x_f at cols 0..511, x_b at 1024..1535

    const int tid  = threadIdx.x;
    const int wave = tid >> 5, lane = tid & 31;
    const int r    = lane & 15;
    const bool hi  = lane >= 16;
    const int m0   = blockIdx.x * 64 + wave * 16;
    const int rowA = m0 + r;
    const int kb   = hi ? 8 : 0;
    const int roff = hi ? 16 : 0;
    const int sc   = tid % XDW;           // staging column (threads < 96 active)
    const int sg   = (tid / XDW) << 4;

    v8f acc[3] = {};
    for (int k0 = 0; k0 < 512; k0 += 32) {
        __syncthreads();
        if (tid < 2 * XDW) {
            #pragma unroll
            for (int it = 0; it < 4; ++it) {
                const int r0 = sg + it * 4;
                float f0 = W[(size_t)(k0 + r0 + 0) * XDW + sc];
                float f1 = W[(size_t)(k0 + r0 + 1) * XDW + sc];
                float f2 = W[(size_t)(k0 + r0 + 2) * XDW + sc];
                float f3 = W[(size_t)(k0 + r0 + 3) * XDW + sc];
                union { v4h v; h2 p[2]; } u;
                u.p[0] = __builtin_amdgcn_cvt_pkrtz(f0, f1);
                u.p[1] = __builtin_amdgcn_cvt_pkrtz(f2, f3);
                *(v4h*)(&bt[sc * BSTR + (((r0 & 8) << 1) | ((r0 & 16) >> 1) | (r0 & 7))]) = u.v;
            }
        }
        __syncthreads();
        const float* ap = A + (size_t)rowA * NXZ + k0 + kb;
        v4f x0 = *(const v4f*)(ap);
        v4f x1 = *(const v4f*)(ap + 4);
        v4f x2 = *(const v4f*)(ap + 16);
        v4f x3 = *(const v4f*)(ap + 20);
        #pragma unroll
        for (int j = 0; j < 4; ++j) {
            x0[j] = siluf(x0[j]); x1[j] = siluf(x1[j]);
            x2[j] = siluf(x2[j]); x3[j] = siluf(x3[j]);
        }
        v16h a = pack_afrag(x0, x1, x2, x3);
        #pragma unroll
        for (int t = 0; t < 3; ++t) {
            v16h bf = *(const v16h*)(&bt[(t * 16 + r) * BSTR + roff]);
            acc[t] = __builtin_amdgcn_wmma_f32_16x16x32_f16(false, a, false, bf,
                                                            (short)0, acc[t], false, false);
        }
    }
    #pragma unroll
    for (int t = 0; t < 3; ++t)
        #pragma unroll
        for (int j = 0; j < 8; ++j)
            XD[(size_t)(m0 + j + (hi ? 8 : 0)) * XDW + t * 16 + r] = acc[t][j];
}

// ---------------------------------------------------------------------------
// Selective scan (fwd + bwd): grid (4 d-chunks, 16 batches, 2 branches).
// Each lane owns channel d; h[16] in registers; 900 sequential steps.
// xd row address is block-uniform -> scalarizable loads, no barriers needed.
// ---------------------------------------------------------------------------
__global__ __launch_bounds__(128) void k_scan(const float* __restrict__ XZ,
                                              const float* __restrict__ XDf,
                                              const float* __restrict__ XDb,
                                              const float* __restrict__ dtWf,
                                              const float* __restrict__ dtWb,
                                              const float* __restrict__ dtBf,
                                              const float* __restrict__ dtBb,
                                              const float* __restrict__ Alf,
                                              const float* __restrict__ Alb,
                                              const float* __restrict__ Dfp,
                                              const float* __restrict__ Dbp,
                                              float* __restrict__ Y) {
    const int d  = blockIdx.x * 128 + threadIdx.x;
    const int b  = blockIdx.y;
    const int br = blockIdx.z;

    const float* XD   = br ? XDb  : XDf;
    const float* dtW  = br ? dtWb : dtWf;
    const float* dtB  = br ? dtBb : dtBf;
    const float* Alog = br ? Alb  : Alf;
    const float* Dp   = br ? Dbp  : Dfp;

    float Ar[NST], w[16], h[NST];
    #pragma unroll
    for (int n = 0; n < NST; ++n) { Ar[n] = -__expf(Alog[d * NST + n]); h[n] = 0.f; }
    #pragma unroll
    for (int k = 0; k < 16; ++k) w[k] = dtW[k * DINNER + d];
    const float bias = dtB[d];
    const float Dd   = Dp[d];
    const int  xcol  = br * 1024 + d;

    for (int ti = 0; ti < LL; ++ti) {
        const int t = br ? (LL - 1 - ti) : ti;
        const size_t row = (size_t)b * LL + t;
        if (ti + 1 < LL) {                       // prefetch next timestep
            const size_t rn = (size_t)b * LL + (br ? (t - 1) : (t + 1));
            __builtin_prefetch(XD + rn * XDW, 0, 3);
            __builtin_prefetch(XZ + rn * NXZ + xcol, 0, 3);
            __builtin_prefetch(XZ + rn * NXZ + xcol + 512, 0, 3);
        }
        const float* xr = XD + row * XDW;        // uniform across block

        const float x = XZ[row * NXZ + xcol];
        const float z = XZ[row * NXZ + xcol + 512];
        const float u = siluf(x);
        float dtr = bias;
        #pragma unroll
        for (int k = 0; k < 16; ++k) dtr += xr[k] * w[k];
        const float dt = softplusf(dtr);
        const float du = dt * u;
        float y = 0.f;
        #pragma unroll
        for (int n = 0; n < NST; ++n) {
            h[n] = __expf(dt * Ar[n]) * h[n] + du * xr[16 + n];
            y   += h[n] * xr[32 + n];
        }
        Y[row * 1024 + br * 512 + d] = (y + u * Dd) * siluf(z);
    }
}

// ---------------------------------------------------------------------------
// Sum over L per (batch, 4 channels): S[16,1024]
// ---------------------------------------------------------------------------
__global__ void k_mean(const float* __restrict__ Y, float* __restrict__ S) {
    const int idx = blockIdx.x * blockDim.x + threadIdx.x;   // 0..4095
    const int b = idx >> 8, c4 = (idx & 255) << 2;
    const float* p = Y + (size_t)b * LL * 1024 + c4;
    v4f s = {};
    for (int t = 0; t < LL; ++t) {
        v4f v = *(const v4f*)(p + (size_t)t * 1024);
        s += v;
    }
    *(v4f*)(S + (size_t)b * 1024 + c4) = s;
}

// ---------------------------------------------------------------------------
// G = sigmoid(mean(y) @ Wg + bg)  — tiny (33 MFLOP), plain VALU
// ---------------------------------------------------------------------------
__global__ void k_gproj(const float* __restrict__ S, const float* __restrict__ W,
                        const float* __restrict__ bias, float* __restrict__ G) {
    const int idx = blockIdx.x * blockDim.x + threadIdx.x;
    const int b = idx >> 10, n = idx & 1023;
    const float* s = S + b * 1024;
    float acc = 0.f;
    for (int k = 0; k < 1024; ++k) acc += s[k] * W[(size_t)k * 1024 + n];
    G[idx] = sigmf(acc * (1.f / (float)LL) + bias[n]);
}

// ---------------------------------------------------------------------------
// GEMM 3: OUT[14400,256] = (Y * g) @ Wout[1024,256]; gate fused into A pack
// ---------------------------------------------------------------------------
__global__ __launch_bounds__(128) void k_out(const float* __restrict__ Y,
                                             const float* __restrict__ G,
                                             const float* __restrict__ W,
                                             float* __restrict__ OUT) {
    __shared__ __attribute__((aligned(32))) _Float16 bt[64 * BSTR];
    const int tid  = threadIdx.x;
    const int wave = tid >> 5, lane = tid & 31;
    const int r    = lane & 15;
    const bool hi  = lane >= 16;
    const int m0   = blockIdx.y * 64 + wave * 16;
    const int n0   = blockIdx.x * 64;
    const int rowA = m0 + r;
    const int kb   = hi ? 8 : 0;
    const int roff = hi ? 16 : 0;
    const int sc   = tid & 63;
    const int sg   = (tid >> 6) << 4;
    const float* Wn   = W + n0;
    const float* grow = G + (size_t)(rowA / LL) * 1024;   // this row's batch

    v8f acc[4] = {};
    for (int k0 = 0; k0 < 1024; k0 += 32) {
        __syncthreads();
        #pragma unroll
        for (int it = 0; it < 4; ++it) {
            const int r0 = sg + it * 4;
            float f0 = Wn[(size_t)(k0 + r0 + 0) * 256 + sc];
            float f1 = Wn[(size_t)(k0 + r0 + 1) * 256 + sc];
            float f2 = Wn[(size_t)(k0 + r0 + 2) * 256 + sc];
            float f3 = Wn[(size_t)(k0 + r0 + 3) * 256 + sc];
            union { v4h v; h2 p[2]; } u;
            u.p[0] = __builtin_amdgcn_cvt_pkrtz(f0, f1);
            u.p[1] = __builtin_amdgcn_cvt_pkrtz(f2, f3);
            *(v4h*)(&bt[sc * BSTR + (((r0 & 8) << 1) | ((r0 & 16) >> 1) | (r0 & 7))]) = u.v;
        }
        __syncthreads();
        const float* ap = Y + (size_t)rowA * 1024 + k0 + kb;
        const float* gp = grow + k0 + kb;
        v4f x0 = *(const v4f*)(ap)      * *(const v4f*)(gp);
        v4f x1 = *(const v4f*)(ap + 4)  * *(const v4f*)(gp + 4);
        v4f x2 = *(const v4f*)(ap + 16) * *(const v4f*)(gp + 16);
        v4f x3 = *(const v4f*)(ap + 20) * *(const v4f*)(gp + 20);
        v16h a = pack_afrag(x0, x1, x2, x3);
        #pragma unroll
        for (int t = 0; t < 4; ++t) {
            v16h bf = *(const v16h*)(&bt[(t * 16 + r) * BSTR + roff]);
            acc[t] = __builtin_amdgcn_wmma_f32_16x16x32_f16(false, a, false, bf,
                                                            (short)0, acc[t], false, false);
        }
    }
    #pragma unroll
    for (int t = 0; t < 4; ++t)
        #pragma unroll
        for (int j = 0; j < 8; ++j)
            OUT[(size_t)(m0 + j + (hi ? 8 : 0)) * 256 + n0 + t * 16 + r] = acc[t][j];
}

// ---------------------------------------------------------------------------
extern "C" void kernel_launch(void* const* d_in, const int* in_sizes, int n_in,
                              void* d_out, int out_size, void* d_ws, size_t ws_size,
                              hipStream_t stream) {
    (void)in_sizes; (void)n_in; (void)out_size; (void)ws_size;
    const float* H    = (const float*)d_in[0];   // hidden_states (16,900,256)
    const float* Win  = (const float*)d_in[1];   // in_proj_w     (256,2048)
    const float* Wxf  = (const float*)d_in[2];   // x_proj_f_w    (512,48)
    const float* Wxb  = (const float*)d_in[3];   // x_proj_b_w    (512,48)
    const float* Wdtf = (const float*)d_in[4];   // dt_proj_f_w   (16,512)
    const float* Wdtb = (const float*)d_in[5];   // dt_proj_b_w   (16,512)
    const float* bdtf = (const float*)d_in[6];   // dt_proj_f_b   (512)
    const float* bdtb = (const float*)d_in[7];   // dt_proj_b_b   (512)
    const float* Alf  = (const float*)d_in[8];   // A_log_f       (512,16)
    const float* Alb  = (const float*)d_in[9];   // A_log_b       (512,16)
    const float* Dfp  = (const float*)d_in[10];  // D_f           (512)
    const float* Dbp  = (const float*)d_in[11];  // D_b           (512)
    const float* Wout = (const float*)d_in[12];  // out_proj_w    (1024,256)
    const float* Wg   = (const float*)d_in[13];  // global_proj_w (1024,1024)
    const float* bg   = (const float*)d_in[14];  // global_proj_b (1024)

    float* ws  = (float*)d_ws;
    float* XZ  = ws;                                   // 14400*2048
    float* XDf = XZ  + (size_t)NROWS * NXZ;            // 14400*48
    float* XDb = XDf + (size_t)NROWS * XDW;            // 14400*48
    float* Yb  = XDb + (size_t)NROWS * XDW;            // 14400*1024
    float* S   = Yb  + (size_t)NROWS * 1024;           // 16*1024
    float* G   = S   + BB * 1024;                      // 16*1024

    k_inproj<<<dim3(2048 / 64, NROWS / 64), 128, 0, stream>>>(H, Win, XZ);
    k_xd    <<<dim3(NROWS / 64, 2),         128, 0, stream>>>(XZ, Wxf, Wxb, XDf, XDb);
    k_scan  <<<dim3(DINNER / 128, BB, 2),   128, 0, stream>>>(XZ, XDf, XDb,
                                                              Wdtf, Wdtb, bdtf, bdtb,
                                                              Alf, Alb, Dfp, Dbp, Yb);
    k_mean  <<<dim3(BB * 1024 / (256 * 4)), 256, 0, stream>>>(Yb, S);
    k_gproj <<<dim3(BB * 1024 / 256),       256, 0, stream>>>(S, Wg, bg, G);
    k_out   <<<dim3(256 / 64, NROWS / 64),  128, 0, stream>>>(Yb, G, Wout, (float*)d_out);
}
